// CausalLocalSelfAttention_61186104099555
// MI455X (gfx1250) — compile-verified, matrix operand order checked by reading, not run on hardware
//
#include <hip/hip_runtime.h>

typedef __bf16 bf16_t;
typedef bf16_t v8bf  __attribute__((ext_vector_type(8)));
typedef bf16_t v16bf __attribute__((ext_vector_type(16)));
typedef float  v8f   __attribute__((ext_vector_type(8)));

// Problem constants (reference: B=2, T=2048, C=1024, H=16, hd=64, window=512)
#define TT   2048
#define CC   1024
#define HH   16
#define HD   64
#define NROW 4096          // B*T
#define N3C  3072          // 3*C
#define WIN  512

// Load a 16x32 bf16 A/B fragment row: lane holds K = {lh*8..lh*8+7, 16+lh*8..+7}
__device__ __forceinline__ v16bf ld_frag(const bf16_t* __restrict__ p, int lh) {
  v8bf lo = *reinterpret_cast<const v8bf*>(p + lh * 8);
  v8bf hi = *reinterpret_cast<const v8bf*>(p + 16 + lh * 8);
  return __builtin_shufflevector(lo, hi, 0,1,2,3,4,5,6,7,8,9,10,11,12,13,14,15);
}

__device__ __forceinline__ v8f wmma_bf16(v16bf a, v16bf b, v8f c) {
  return __builtin_amdgcn_wmma_f32_16x16x32_bf16(false, a, false, b, (short)0, c,
                                                 false, false);
}

__global__ __launch_bounds__(256) void cvt_bf16_k(const float* __restrict__ in,
                                                  bf16_t* __restrict__ out, int n) {
  int i = blockIdx.x * 256 + threadIdx.x;
  if (i < n) out[i] = (bf16_t)in[i];
}

// ---------------- QKV GEMM (64x64 per wave, 4x4 WMMA blocking) + fused RoPE ----
// tiles = (4096/64)*(3072/64) = 3072 waves; 128-thr blocks (4 waves) -> 768 blocks
__global__ __launch_bounds__(128) void qkv_rope_k(const bf16_t* __restrict__ xb,
                                                  const bf16_t* __restrict__ wb,
                                                  bf16_t* __restrict__ qo,
                                                  bf16_t* __restrict__ ko,
                                                  bf16_t* __restrict__ vt) {
  const int lane = threadIdx.x & 31;
  const int lh = lane >> 4, ln = lane & 15;
  const int tile = blockIdx.x * 4 + (threadIdx.x >> 5);
  const int mt = tile / (N3C / 64), nt = tile % (N3C / 64);
  const int r0 = mt * 64, n0 = nt * 64;

  const bf16_t* arow[4];
  const bf16_t* brow[4];
  #pragma unroll
  for (int i = 0; i < 4; ++i) arow[i] = xb + (size_t)(r0 + i * 16 + ln) * CC;
  #pragma unroll
  for (int j = 0; j < 4; ++j) brow[j] = wb + (size_t)(n0 + j * 16 + ln) * CC;

  v8f acc[4][4];
  const v8f vz = {0.f,0.f,0.f,0.f,0.f,0.f,0.f,0.f};
  #pragma unroll
  for (int i = 0; i < 4; ++i)
    #pragma unroll
    for (int j = 0; j < 4; ++j) acc[i][j] = vz;

  for (int kk = 0; kk < CC; kk += 32) {
    v16bf af[4], bfr[4];
    #pragma unroll
    for (int i = 0; i < 4; ++i) af[i] = ld_frag(arow[i] + kk, lh);
    #pragma unroll
    for (int j = 0; j < 4; ++j) bfr[j] = ld_frag(brow[j] + kk, lh);
    #pragma unroll
    for (int i = 0; i < 4; ++i)
      #pragma unroll
      for (int j = 0; j < 4; ++j)
        acc[i][j] = wmma_bf16(af[i], bfr[j], acc[i][j]);  // 16 independent WMMAs
  }

  const int b = r0 >> 11, t0 = r0 & (TT - 1);
  const int which = n0 >> 10;                 // n0 64-aligned -> q/k/v constant
  const int h = (n0 & (CC - 1)) >> 6;         // 64-wide tile = exactly one head

  if (which == 2) {                           // V: store transposed [B,H,hd,T]
    #pragma unroll
    for (int j = 0; j < 4; ++j) {
      const int d = j * 16 + ln;
      bf16_t* dst = vt + ((size_t)(b * HH + h) * HD + d) * TT + t0;
      #pragma unroll
      for (int i = 0; i < 4; ++i)
        #pragma unroll
        for (int v = 0; v < 8; ++v)
          dst[i * 16 + v + 8 * lh] = (bf16_t)acc[i][j][v];
    }
  } else {                                    // Q/K: RoPE then [B,H,T,hd]
    bf16_t* base = (which == 0 ? qo : ko) + ((size_t)(b * HH + h) * TT + t0) * HD;
    #pragma unroll
    for (int j = 0; j < 4; ++j) {
      const int d = j * 16 + ln;
      // inv_freq = 10000^{-(d/2)/32} = 2^{-(log2(1e4)/32) * (d>>1)}
      const float invf = exp2f(-0.4152410118f * (float)(d >> 1));
      #pragma unroll
      for (int i = 0; i < 4; ++i)
        #pragma unroll
        for (int v = 0; v < 8; ++v) {
          const int m = i * 16 + v + 8 * lh;
          float s, c;
          __sincosf((float)(t0 + m) * invf, &s, &c);
          float val = acc[i][j][v];
          float prt = __shfl_xor(val, 1, 32);           // interleaved pair partner
          float o = (d & 1) ? fmaf(val, c, prt * s)     // x_odd*cos + x_even*sin
                            : fmaf(val, c, -prt * s);   // x_even*cos - x_odd*sin
          base[(size_t)m * HD + d] = (bf16_t)o;
        }
    }
  }
}

// ---------------- Local causal flash attention ----------------
// grid: 512 blocks x 256 thr = 4096 waves = B*H*(T/16)
__global__ __launch_bounds__(256) void attn_k(const bf16_t* __restrict__ q,
                                              const bf16_t* __restrict__ kmat,
                                              const bf16_t* __restrict__ vt,
                                              bf16_t* __restrict__ y) {
  __shared__ alignas(16) bf16_t plds[8][16 * 32];     // per-wave P staging (1 KB)
  const int lane = threadIdx.x & 31;
  const int lh = lane >> 4, ln = lane & 15;
  const int wid = threadIdx.x >> 5;
  const int w = blockIdx.x * 8 + wid;
  const int qt0 = (w & 127) * 16;
  const int bh = w >> 7;

  const bf16_t* qb = q    + (size_t)bh * TT * HD;
  const bf16_t* kb = kmat + (size_t)bh * TT * HD;
  const bf16_t* vb = vt   + (size_t)bh * HD * TT;

  const bf16_t* qrow = qb + (size_t)(qt0 + ln) * HD;
  v16bf qa0 = ld_frag(qrow, lh), qa1 = ld_frag(qrow + 32, lh);

  float mrow[8], lrow[8];
  v8f accd[4];
  const v8f vz = {0.f,0.f,0.f,0.f,0.f,0.f,0.f,0.f};
  #pragma unroll
  for (int i = 0; i < 8; ++i) { mrow[i] = -1e30f; lrow[i] = 0.f; }
  #pragma unroll
  for (int i = 0; i < 4; ++i) accd[i] = vz;

  int kstart = qt0 - WIN; if (kstart < 0) kstart = 0;
  kstart &= ~31;                                      // 32-aligned; kt0 <= 2016 always
  for (int kt0 = kstart; kt0 <= qt0 + 15; kt0 += 32) {
    const bf16_t* kr0 = kb + (size_t)(kt0 + ln) * HD;
    const bf16_t* kr1 = kb + (size_t)(kt0 + 16 + ln) * HD;
    v8f s0 = wmma_bf16(qa1, ld_frag(kr0 + 32, lh),
                       wmma_bf16(qa0, ld_frag(kr0, lh), vz));
    v8f s1 = wmma_bf16(qa1, ld_frag(kr1 + 32, lh),
                       wmma_bf16(qa0, ld_frag(kr1, lh), vz));

    #pragma unroll
    for (int v = 0; v < 8; ++v) {                     // scale + causal/window mask
      const int qtg = qt0 + v + 8 * lh;
      const int kg0 = kt0 + ln, kg1 = kg0 + 16;
      float a = s0[v] * 0.125f, b2 = s1[v] * 0.125f;  // 1/sqrt(64)
      s0[v] = (kg0 <= qtg && qtg - kg0 <= WIN) ? a  : -1e30f;
      s1[v] = (kg1 <= qtg && qtg - kg1 <= WIN) ? b2 : -1e30f;
    }
    #pragma unroll
    for (int v = 0; v < 8; ++v) {                     // online softmax per row
      float mt2 = fmaxf(s0[v], s1[v]);
      #pragma unroll
      for (int off = 1; off < 16; off <<= 1)
        mt2 = fmaxf(mt2, __shfl_xor(mt2, off, 32));
      float mnew = fmaxf(mrow[v], mt2);
      float alpha = __expf(mrow[v] - mnew);
      mrow[v] = mnew;
      float p0 = __expf(s0[v] - mnew);
      float p1 = __expf(s1[v] - mnew);
      float rs = p0 + p1;
      #pragma unroll
      for (int off = 1; off < 16; off <<= 1)
        rs += __shfl_xor(rs, off, 32);
      lrow[v] = lrow[v] * alpha + rs;
      #pragma unroll
      for (int nd = 0; nd < 4; ++nd) accd[nd][v] *= alpha;
      const int m = v + 8 * lh;                       // D-layout -> LDS transpose
      plds[wid][m * 32 + ln]      = (bf16_t)p0;
      plds[wid][m * 32 + 16 + ln] = (bf16_t)p1;
    }
    asm volatile("s_wait_dscnt 0" ::: "memory");
    v16bf pa = ld_frag(&plds[wid][ln * 32], lh);      // P as A-fragment (16x32)
    #pragma unroll
    for (int nd = 0; nd < 4; ++nd) {                  // acc += P @ V (vT rows contiguous in key)
      const bf16_t* vrow = vb + (size_t)(nd * 16 + ln) * TT + kt0;
      accd[nd] = wmma_bf16(pa, ld_frag(vrow, lh), accd[nd]);
    }
  }

  const int b = bh >> 4, h = bh & 15;
  #pragma unroll
  for (int nd = 0; nd < 4; ++nd) {
    bf16_t* yrow = y + (size_t)(b * TT + qt0) * CC + h * HD + nd * 16 + ln;
    #pragma unroll
    for (int v = 0; v < 8; ++v) {
      const int m = v + 8 * lh;
      yrow[(size_t)m * CC] = (bf16_t)(accd[nd][v] / lrow[v]);
    }
  }
}

// ---------------- Output projection GEMM (64x64 per wave) + bias ----------------
// tiles = (4096/64)*(1024/64) = 1024 waves; 128-thr blocks -> 256 blocks
__global__ __launch_bounds__(128) void proj_k(const bf16_t* __restrict__ yb,
                                              const bf16_t* __restrict__ wb,
                                              const float* __restrict__ bias,
                                              float* __restrict__ out) {
  const int lane = threadIdx.x & 31;
  const int lh = lane >> 4, ln = lane & 15;
  const int tile = blockIdx.x * 4 + (threadIdx.x >> 5);
  const int mt = tile / (CC / 64), nt = tile % (CC / 64);
  const int r0 = mt * 64, n0 = nt * 64;

  const bf16_t* arow[4];
  const bf16_t* brow[4];
  #pragma unroll
  for (int i = 0; i < 4; ++i) arow[i] = yb + (size_t)(r0 + i * 16 + ln) * CC;
  #pragma unroll
  for (int j = 0; j < 4; ++j) brow[j] = wb + (size_t)(n0 + j * 16 + ln) * CC;

  v8f acc[4][4];
  const v8f vz = {0.f,0.f,0.f,0.f,0.f,0.f,0.f,0.f};
  #pragma unroll
  for (int i = 0; i < 4; ++i)
    #pragma unroll
    for (int j = 0; j < 4; ++j) acc[i][j] = vz;

  for (int kk = 0; kk < CC; kk += 32) {
    v16bf af[4], bfr[4];
    #pragma unroll
    for (int i = 0; i < 4; ++i) af[i] = ld_frag(arow[i] + kk, lh);
    #pragma unroll
    for (int j = 0; j < 4; ++j) bfr[j] = ld_frag(brow[j] + kk, lh);
    #pragma unroll
    for (int i = 0; i < 4; ++i)
      #pragma unroll
      for (int j = 0; j < 4; ++j)
        acc[i][j] = wmma_bf16(af[i], bfr[j], acc[i][j]);
  }

  #pragma unroll
  for (int j = 0; j < 4; ++j) {
    const float bv = bias[n0 + j * 16 + ln];
    #pragma unroll
    for (int i = 0; i < 4; ++i)
      #pragma unroll
      for (int v = 0; v < 8; ++v) {
        const int m = i * 16 + v + 8 * lh;
        out[(size_t)(r0 + m) * CC + n0 + j * 16 + ln] = acc[i][j][v] + bv;
      }
  }
}

extern "C" void kernel_launch(void* const* d_in, const int* in_sizes, int n_in,
                              void* d_out, int out_size, void* d_ws, size_t ws_size,
                              hipStream_t stream) {
  (void)in_sizes; (void)n_in; (void)out_size; (void)ws_size;
  const float* x     = (const float*)d_in[0];   // [B,T,C]
  const float* Wqkv  = (const float*)d_in[1];   // [3C,C]
  const float* Wproj = (const float*)d_in[2];   // [C,C]
  const float* bproj = (const float*)d_in[3];   // [C]

  char* ws = (char*)d_ws;                       // all offsets 16B-aligned
  bf16_t* xb     = (bf16_t*)(ws + 0);           //  8 MB  x bf16
  bf16_t* wqkvb  = (bf16_t*)(ws + 8388608);     //  6 MB
  bf16_t* wprojb = (bf16_t*)(ws + 14680064);    //  2 MB
  bf16_t* qb     = (bf16_t*)(ws + 16777216);    //  8 MB  [B,H,T,hd]
  bf16_t* kbuf   = (bf16_t*)(ws + 25165824);    //  8 MB  [B,H,T,hd]
  bf16_t* vt     = (bf16_t*)(ws + 33554432);    //  8 MB  [B,H,hd,T] (+4KB pad)
  bf16_t* yb     = (bf16_t*)(ws + 41947136);    //  8 MB  [B,T,C]

  cvt_bf16_k<<<NROW * CC / 256, 256, 0, stream>>>(x, xb, NROW * CC);
  cvt_bf16_k<<<N3C * CC / 256, 256, 0, stream>>>(Wqkv, wqkvb, N3C * CC);
  cvt_bf16_k<<<CC * CC / 256, 256, 0, stream>>>(Wproj, wprojb, CC * CC);

  qkv_rope_k<<<(NROW / 64) * (N3C / 64) / 4, 128, 0, stream>>>(xb, wqkvb, qb, kbuf, vt);
  attn_k<<<2 * HH * (TT / 16) / 8, 256, 0, stream>>>(qb, kbuf, vt, yb);
  proj_k<<<(NROW / 64) * (CC / 64) / 4, 128, 0, stream>>>(yb, wprojb, bproj,
                                                          (float*)d_out);
}
